// RelativeEncoderLayer_74268574482839
// MI455X (gfx1250) — matrix operand order repaired and checked by reading, not compiled, not run on hardware
//
#include <hip/hip_runtime.h>

#define D_MODEL 1024
#define N_HEAD  16
#define D_KH    64
#define D_INNER 4096
#define BATCH   8
#define SEQ     1024
#define NREL    33
#define MTOT    (BATCH*SEQ)

typedef __bf16 bf16;
typedef __attribute__((ext_vector_type(16))) __bf16 v16bf;
typedef __attribute__((ext_vector_type(8)))  __bf16 v8bf;
typedef __attribute__((ext_vector_type(8)))  float  v8f;

__device__ __forceinline__ bf16 f2bf(float f) {
  unsigned u = __builtin_bit_cast(unsigned, f);
  unsigned r = u + 0x7FFFu + ((u >> 16) & 1u);
  unsigned short s = (unsigned short)(r >> 16);
  return __builtin_bit_cast(bf16, s);
}
__device__ __forceinline__ float bf2f(bf16 h) {
  unsigned short s = __builtin_bit_cast(unsigned short, h);
  unsigned u = ((unsigned)s) << 16;
  return __builtin_bit_cast(float, u);
}

union AFrag { v16bf v; v8bf h[2]; };

// ---- gfx1250 async global->LDS path (ASYNCcnt), guarded by builtin probe ----
#if defined(__has_builtin)
#  if __has_builtin(__builtin_amdgcn_global_load_async_to_lds_b128) && \
      __has_builtin(__builtin_amdgcn_s_wait_asynccnt)
#    define HAVE_ASYNC_LDS 1
#  endif
#endif

#ifdef HAVE_ASYNC_LDS
typedef int v4i __attribute__((ext_vector_type(4)));
typedef __attribute__((address_space(1))) v4i gv4i;
typedef __attribute__((address_space(3))) v4i lv4i;
__device__ __forceinline__ void async_cp16(const void* g, void* l) {
  __builtin_amdgcn_global_load_async_to_lds_b128((gv4i*)g, (lv4i*)l, 0, 0);
}
#endif

// ---------------------------------------------------------------------------
// Generic bf16 WMMA GEMM, double-buffered LDS.  C = epi(A @ B [^T]).
// EPI: 0 = f32 out, 1 = bf16 out, 2 = relu->bf16, 3 = +bias+Res -> f32,
//      4 = score: v*scale + P[q, clip(k-q)] -> f32
// BT : 0 = B is [K,N] row-major,  1 = B is [N,K] row-major (compute A @ B^T)
// mode per operand: 0 -> off = z*s1 ;  1 -> bh=bh0+z, off=(bh>>4)*s1+(bh&15)*s2
// ---------------------------------------------------------------------------
template<int EPI, int BT>
__global__ __launch_bounds__(256) void gemm_wmma(
    const bf16* __restrict__ A, int lda, long sA1, long sA2, int amode,
    const bf16* __restrict__ B, int ldb, long sB1, long sB2, int bmode,
    void* __restrict__ C, int ldc, long sC1, long sC2, int cmode,
    const float* __restrict__ bias,
    const float* __restrict__ Res,
    const float* __restrict__ P, long sP,
    int M, int N, int K, float scale, int bh0)
{
  __shared__ __align__(16) unsigned short sa[2][128 * 40];
  __shared__ __align__(16) unsigned short sbuf[2][128 * 40];

  const int tid  = threadIdx.x;
  const int lane = tid & 31;
  const int wid  = tid >> 5;
  const int wy   = wid >> 2;     // 0..1  -> 64 rows each
  const int wx   = wid & 3;      // 0..3  -> 32 cols each
  const int m0   = blockIdx.y * 128;
  const int n0   = blockIdx.x * 128;
  const int z    = blockIdx.z;
  const int bh   = bh0 + z;

  const long offA = amode ? ((long)(bh >> 4) * sA1 + (long)(bh & 15) * sA2)
                          : (long)z * sA1;
  const long offB = bmode ? ((long)(bh >> 4) * sB1 + (long)(bh & 15) * sB2)
                          : (long)z * sB1;
  const long offC = cmode ? ((long)(bh >> 4) * sC1 + (long)(bh & 15) * sC2)
                          : (long)z * sC1;
  const long offP = (long)bh * sP;

  const bf16* Ab = A + offA;
  const bf16* Bb = B + offB;
  const int nk = K / 32;

  // ---- tile loader: global -> LDS buffer `buf` for K-step `kt`
  auto loadTiles = [&](int kt, int buf) {
    const int k0 = kt * 32;
    // A tile: 128x32 bf16, 512 x 16B slots, 2 per thread
#pragma unroll
    for (int i = 0; i < 2; ++i) {
      int idx = tid + i * 256;
      int row = idx >> 2, seg = idx & 3;
      int gm = m0 + row, gk = k0 + seg * 8;
#ifdef HAVE_ASYNC_LDS
      if (gm < M)
        async_cp16(Ab + (long)gm * lda + gk, &sa[buf][row * 40 + seg * 8]);
#else
      uint4 val = {0u, 0u, 0u, 0u};
      if (gm < M) val = *(const uint4*)(Ab + (long)gm * lda + gk);
      *(uint4*)(&sa[buf][row * 40 + seg * 8]) = val;
#endif
    }
    // B tile -> LDS as [n][k] (lanes later read two 8-K runs, at ks and ks+16)
    if (BT) {
#pragma unroll
      for (int i = 0; i < 2; ++i) {
        int idx = tid + i * 256;
        int nr = idx >> 2, seg = idx & 3;
        int gn = n0 + nr, gk = k0 + seg * 8;
#ifdef HAVE_ASYNC_LDS
        if (gn < N)
          async_cp16(Bb + (long)gn * ldb + gk, &sbuf[buf][nr * 40 + seg * 8]);
#else
        uint4 val = {0u, 0u, 0u, 0u};
        if (gn < N) val = *(const uint4*)(Bb + (long)gn * ldb + gk);
        *(uint4*)(&sbuf[buf][nr * 40 + seg * 8]) = val;
#endif
      }
    } else {
      // row-major B: transpose through VGPRs (no TR instruction assumed)
#pragma unroll
      for (int i = 0; i < 2; ++i) {
        int idx = tid + i * 256;
        int kr = idx >> 4, seg = idx & 15;
        int gk = k0 + kr, gn = n0 + seg * 8;
        union { uint4 u; unsigned short s[8]; } tv;
        tv.u = (uint4){0u, 0u, 0u, 0u};
        if (gn + 7 < N) tv.u = *(const uint4*)(Bb + (long)gk * ldb + gn);
#pragma unroll
        for (int e = 0; e < 8; ++e)
          sbuf[buf][(seg * 8 + e) * 40 + kr] = tv.s[e];
      }
    }
    // L2 prefetch of the next tile (global_prefetch_b8)
    if (kt + 1 < nk) {
      int row = tid >> 2, seg = tid & 3;
      int gm = m0 + row;
      if (gm < M)
        __builtin_prefetch(Ab + (long)gm * lda + (kt + 1) * 32 + seg * 8, 0, 1);
    }
  };

  v8f acc[4][2];
#pragma unroll
  for (int a = 0; a < 4; ++a)
#pragma unroll
    for (int c = 0; c < 2; ++c)
      acc[a][c] = (v8f){0.f, 0.f, 0.f, 0.f, 0.f, 0.f, 0.f, 0.f};

  loadTiles(0, 0);
  int cur = 0;

  for (int kt = 0; kt < nk; ++kt) {
#ifdef HAVE_ASYNC_LDS
    __builtin_amdgcn_s_wait_asynccnt(0);   // my async tile writes done
#endif
    __syncthreads();                       // everyone's tile visible / prev reads done
    if (kt + 1 < nk) loadTiles(kt + 1, cur ^ 1);  // overlap next loads with WMMA

    // ---- fragments (ISA 16-bit A 16x32 layout / B same table-K layout)
    AFrag af[4], bfr[2];
    const int kb = (lane >> 4) * 8;
#pragma unroll
    for (int a = 0; a < 4; ++a) {
      int r = wy * 64 + a * 16 + (lane & 15);
      af[a].h[0] = *(const v8bf*)(&sa[cur][r * 40 + kb]);
      af[a].h[1] = *(const v8bf*)(&sa[cur][r * 40 + kb + 16]);
    }
    const int ks = (lane >> 4) * 8;
#pragma unroll
    for (int c = 0; c < 2; ++c) {
      int n = wx * 32 + c * 16 + (lane & 15);
      bfr[c].h[0] = *(const v8bf*)(&sbuf[cur][n * 40 + ks]);
      bfr[c].h[1] = *(const v8bf*)(&sbuf[cur][n * 40 + ks + 16]);
    }
#pragma unroll
    for (int a = 0; a < 4; ++a)
#pragma unroll
      for (int c = 0; c < 2; ++c)
        acc[a][c] = __builtin_amdgcn_wmma_f32_16x16x32_bf16(
            false, af[a].v, false, bfr[c].v, (short)0, acc[a][c], false, false);

    cur ^= 1;
  }

  // ---- epilogue (C layout: lane l, vgpr r -> m = r + (l>>4)*8, n = l&15)
#pragma unroll
  for (int a = 0; a < 4; ++a) {
#pragma unroll
    for (int c = 0; c < 2; ++c) {
#pragma unroll
      for (int r = 0; r < 8; ++r) {
        int gm = m0 + wy * 64 + a * 16 + r + ((lane >> 4) * 8);
        int gn = n0 + wx * 32 + c * 16 + (lane & 15);
        if (gm < M && gn < N) {
          float v = acc[a][c][r];
          if (bias) v += bias[gn];
          if (EPI == 3) v += Res[(long)gm * ldc + gn];
          if (EPI == 4) {
            int d = gn - gm;
            d = (d < -16) ? -16 : (d > 16 ? 16 : d);
            v = v * scale + P[offP + (long)gm * NREL + d + 16];
          }
          if (EPI == 2) v = fmaxf(v, 0.f);
          if (EPI == 0 || EPI == 3 || EPI == 4)
            ((float*)C)[offC + (long)gm * ldc + gn] = v;
          else
            ((bf16*)C)[offC + (long)gm * ldc + gn] = f2bf(v);
        }
      }
    }
  }
}

// ---------------------------------------------------------------------------
__global__ void cvt_f32_bf16(const float* __restrict__ src,
                             bf16* __restrict__ dst, long n) {
  long i = (long)blockIdx.x * blockDim.x + threadIdx.x;
  long stride = (long)gridDim.x * blockDim.x;
  for (; i < n; i += stride) dst[i] = f2bf(src[i]);
}

// P[bh,q,r] = Qh[b,q,h,:] . rel_k[r,:]
__global__ void relk_proj(const bf16* __restrict__ Qh,
                          const float* __restrict__ rel_k,
                          float* __restrict__ P) {
  int q = blockIdx.x, bh = blockIdx.y;
  int b = bh >> 4, h = bh & 15;
  int r = threadIdx.x;
  if (r >= NREL) return;
  const bf16*  qrow = Qh + ((long)b * SEQ + q) * D_MODEL + h * D_KH;
  const float* krow = rel_k + r * D_KH;
  float acc = 0.f;
#pragma unroll 8
  for (int d = 0; d < D_KH; ++d) acc += bf2f(qrow[d]) * krow[d];
  P[((long)bh * SEQ + q) * NREL + r] = acc;
}

// row softmax + 33-bucket relative sums (ds_add_f32 LDS atomics)
__global__ __launch_bounds__(256) void softmax_rel(
    const float* __restrict__ score, bf16* __restrict__ attn,
    float* __restrict__ Sb, int bh0) {
  int q = blockIdx.x, z = blockIdx.y;
  long rowoff = ((long)z * SEQ + q) * SEQ;
  const float* s = score + rowoff;
  bf16* o = attn + rowoff;
  __shared__ float red[256];
  __shared__ float bins[NREL];
  int tid = threadIdx.x;
  if (tid < NREL) bins[tid] = 0.f;

  float m = -1e30f;
  for (int k = tid; k < SEQ; k += 256) m = fmaxf(m, s[k]);
  red[tid] = m; __syncthreads();
  for (int st = 128; st > 0; st >>= 1) {
    if (tid < st) red[tid] = fmaxf(red[tid], red[tid + st]);
    __syncthreads();
  }
  m = red[0]; __syncthreads();

  float sum = 0.f;
  for (int k = tid; k < SEQ; k += 256) sum += __expf(s[k] - m);
  red[tid] = sum; __syncthreads();
  for (int st = 128; st > 0; st >>= 1) {
    if (tid < st) red[tid] += red[tid + st];
    __syncthreads();
  }
  float inv = 1.f / red[0];

  for (int k = tid; k < SEQ; k += 256) {
    float p = __expf(s[k] - m) * inv;
    o[k] = f2bf(p);
    int d = k - q;
    d = (d < -16) ? -16 : (d > 16 ? 16 : d);
    atomicAdd(&bins[d + 16], p);
  }
  __syncthreads();
  if (tid < NREL)
    Sb[(((long)(bh0 + z)) * SEQ + q) * NREL + tid] = bins[tid];
}

// ctx[b,q,h*64+d] += sum_r Sb[bh,q,r] * rel_v[r,d]
__global__ void addw2(bf16* __restrict__ ctx, const float* __restrict__ Sb,
                      const float* __restrict__ rel_v) {
  int q = blockIdx.x, bh = blockIdx.y;
  int b = bh >> 4, h = bh & 15;
  int d = threadIdx.x;  // 64
  const float* sb = Sb + ((long)bh * SEQ + q) * NREL;
  float acc = 0.f;
#pragma unroll
  for (int r = 0; r < NREL; ++r) acc += sb[r] * rel_v[r * D_KH + d];
  long off = ((long)b * SEQ + q) * D_MODEL + h * D_KH + d;
  ctx[off] = f2bf(bf2f(ctx[off]) + acc);
}

__global__ __launch_bounds__(256) void layernorm_k(
    const float* __restrict__ x, float* __restrict__ yf,
    bf16* __restrict__ ybf, const float* __restrict__ g,
    const float* __restrict__ bta) {
  long row = blockIdx.x;
  const float* xr = x + row * D_MODEL;
  __shared__ float r1[256], r2[256];
  int tid = threadIdx.x;
  float s = 0.f, s2 = 0.f;
  for (int i = tid; i < D_MODEL; i += 256) {
    float v = xr[i]; s += v; s2 += v * v;
  }
  r1[tid] = s; r2[tid] = s2; __syncthreads();
  for (int st = 128; st > 0; st >>= 1) {
    if (tid < st) { r1[tid] += r1[tid + st]; r2[tid] += r2[tid + st]; }
    __syncthreads();
  }
  float mean = r1[0] * (1.f / D_MODEL);
  float var  = r2[0] * (1.f / D_MODEL) - mean * mean;
  float inv  = rsqrtf(var + 1e-6f);
  for (int i = tid; i < D_MODEL; i += 256) {
    float v = (xr[i] - mean) * inv * g[i] + bta[i];
    if (yf)  yf[row * D_MODEL + i] = v;
    if (ybf) ybf[row * D_MODEL + i] = f2bf(v);
  }
}

// ---------------------------------------------------------------------------
extern "C" void kernel_launch(void* const* d_in, const int* in_sizes, int n_in,
                              void* d_out, int out_size, void* d_ws, size_t ws_size,
                              hipStream_t stream) {
  (void)in_sizes; (void)n_in; (void)out_size; (void)ws_size;
  const float* q_in  = (const float*)d_in[0];
  const float* k_in  = (const float*)d_in[1];
  const float* v_in  = (const float*)d_in[2];
  const float* wq  = (const float*)d_in[3];  const float* bq  = (const float*)d_in[4];
  const float* wk  = (const float*)d_in[5];  const float* bk  = (const float*)d_in[6];
  const float* wv  = (const float*)d_in[7];  const float* bv  = (const float*)d_in[8];
  const float* wfc = (const float*)d_in[9];  const float* bfc = (const float*)d_in[10];
  const float* w1  = (const float*)d_in[11]; const float* b1  = (const float*)d_in[12];
  const float* w2  = (const float*)d_in[13]; const float* b2  = (const float*)d_in[14];
  const float* ln_g = (const float*)d_in[15];
  const float* ln_b = (const float*)d_in[16];
  const float* rel_k = (const float*)d_in[17];
  const float* rel_v = (const float*)d_in[18];

  // ---- workspace bump allocator
  char* base = (char*)d_ws;
  size_t o = 0;
  auto alloc = [&](size_t bytes) {
    size_t r = o; o = (o + bytes + 255) & ~(size_t)255; return r;
  };
  const long ME = (long)MTOT * D_MODEL;          // 8192*1024
  bf16* Xq  = (bf16*)(base + alloc(ME * 2));
  bf16* Xk  = (bf16*)(base + alloc(ME * 2));
  bf16* Xv  = (bf16*)(base + alloc(ME * 2));
  bf16* Wq  = (bf16*)(base + alloc((long)D_MODEL * D_MODEL * 2));
  bf16* Wk  = (bf16*)(base + alloc((long)D_MODEL * D_MODEL * 2));
  bf16* Wv  = (bf16*)(base + alloc((long)D_MODEL * D_MODEL * 2));
  bf16* Wfc = (bf16*)(base + alloc((long)D_MODEL * D_MODEL * 2));
  bf16* W1  = (bf16*)(base + alloc((long)D_MODEL * D_INNER * 2));
  bf16* W2  = (bf16*)(base + alloc((long)D_INNER * D_MODEL * 2));
  bf16* Qh  = (bf16*)(base + alloc(ME * 2));
  bf16* Kh  = (bf16*)(base + alloc(ME * 2));
  bf16* Vh  = (bf16*)(base + alloc(ME * 2));
  float* Pb = (float*)(base + alloc((long)BATCH * N_HEAD * SEQ * NREL * 4));
  float* Sb = (float*)(base + alloc((long)BATCH * N_HEAD * SEQ * NREL * 4));
  const int CH = 16;                              // bh chunk
  float* score = (float*)(base + alloc((long)CH * SEQ * SEQ * 4));
  bf16*  attnb = (bf16*)(base + alloc((long)CH * SEQ * SEQ * 2));
  bf16*  ctx   = (bf16*)(base + alloc(ME * 2));
  float* y1    = (float*)(base + alloc(ME * 4));
  float* ln1f  = (float*)(base + alloc(ME * 4));
  bf16*  ln1b  = (bf16*)(base + alloc(ME * 2));
  bf16*  hid   = (bf16*)(base + alloc((long)MTOT * D_INNER * 2));
  float* y2    = (float*)(base + alloc(ME * 4));

  // ---- 1. convert inputs + weights to bf16
  cvt_f32_bf16<<<8192, 256, 0, stream>>>(q_in, Xq, ME);
  cvt_f32_bf16<<<8192, 256, 0, stream>>>(k_in, Xk, ME);
  cvt_f32_bf16<<<8192, 256, 0, stream>>>(v_in, Xv, ME);
  cvt_f32_bf16<<<2048, 256, 0, stream>>>(wq, Wq, (long)D_MODEL * D_MODEL);
  cvt_f32_bf16<<<2048, 256, 0, stream>>>(wk, Wk, (long)D_MODEL * D_MODEL);
  cvt_f32_bf16<<<2048, 256, 0, stream>>>(wv, Wv, (long)D_MODEL * D_MODEL);
  cvt_f32_bf16<<<2048, 256, 0, stream>>>(wfc, Wfc, (long)D_MODEL * D_MODEL);
  cvt_f32_bf16<<<8192, 256, 0, stream>>>(w1, W1, (long)D_MODEL * D_INNER);
  cvt_f32_bf16<<<8192, 256, 0, stream>>>(w2, W2, (long)D_INNER * D_MODEL);

  // ---- 2. Q/K/V projections (bf16 out == head-major layout by construction)
  dim3 gp(D_MODEL / 128, MTOT / 128, 1);
  gemm_wmma<1, 0><<<gp, 256, 0, stream>>>(Xq, D_MODEL, 0, 0, 0,
      Wq, D_MODEL, 0, 0, 0, Qh, D_MODEL, 0, 0, 0,
      bq, nullptr, nullptr, 0, MTOT, D_MODEL, D_MODEL, 1.f, 0);
  gemm_wmma<1, 0><<<gp, 256, 0, stream>>>(Xk, D_MODEL, 0, 0, 0,
      Wk, D_MODEL, 0, 0, 0, Kh, D_MODEL, 0, 0, 0,
      bk, nullptr, nullptr, 0, MTOT, D_MODEL, D_MODEL, 1.f, 0);
  gemm_wmma<1, 0><<<gp, 256, 0, stream>>>(Xv, D_MODEL, 0, 0, 0,
      Wv, D_MODEL, 0, 0, 0, Vh, D_MODEL, 0, 0, 0,
      bv, nullptr, nullptr, 0, MTOT, D_MODEL, D_MODEL, 1.f, 0);

  // ---- 3. P = Qh @ rel_k^T  (33 rel rows)
  relk_proj<<<dim3(SEQ, BATCH * N_HEAD), 64, 0, stream>>>(Qh, rel_k, Pb);

  // ---- 4. attention, chunked 16 (b,h) slices at a time
  const long sBH_A = (long)SEQ * D_MODEL;  // per-batch stride into head tensors
  for (int c = 0; c < (BATCH * N_HEAD) / CH; ++c) {
    int bh0 = c * CH;
    // scores = Qh @ Kh^T * (1/8) + P-gather
    gemm_wmma<4, 1><<<dim3(SEQ / 128, SEQ / 128, CH), 256, 0, stream>>>(
        Qh, D_MODEL, sBH_A, D_KH, 1,
        Kh, D_MODEL, sBH_A, D_KH, 1,
        score, SEQ, (long)SEQ * SEQ, 0, 0,
        nullptr, nullptr, Pb, (long)SEQ * NREL,
        SEQ, SEQ, D_KH, 0.125f, bh0);
    softmax_rel<<<dim3(SEQ, CH), 256, 0, stream>>>(score, attnb, Sb, bh0);
    // ctx slice = attn @ Vh
    gemm_wmma<1, 0><<<dim3(1, SEQ / 128, CH), 256, 0, stream>>>(
        attnb, SEQ, (long)SEQ * SEQ, 0, 0,
        Vh, D_MODEL, sBH_A, D_KH, 1,
        ctx, D_MODEL, sBH_A, D_KH, 1,
        nullptr, nullptr, nullptr, 0,
        SEQ, D_KH, SEQ, 1.f, bh0);
  }

  // ---- 5. add relative-value term
  addw2<<<dim3(SEQ, BATCH * N_HEAD), 64, 0, stream>>>(ctx, Sb, rel_v);

  // ---- 6. out-proj + residual, LN1
  gemm_wmma<3, 0><<<gp, 256, 0, stream>>>(ctx, D_MODEL, 0, 0, 0,
      Wfc, D_MODEL, 0, 0, 0, y1, D_MODEL, 0, 0, 0,
      bfc, q_in, nullptr, 0, MTOT, D_MODEL, D_MODEL, 1.f, 0);
  layernorm_k<<<MTOT, 256, 0, stream>>>(y1, ln1f, ln1b, ln_g, ln_b);

  // ---- 7. FFN
  gemm_wmma<2, 0><<<dim3(D_INNER / 128, MTOT / 128, 1), 256, 0, stream>>>(
      ln1b, D_MODEL, 0, 0, 0, W1, D_INNER, 0, 0, 0, hid, D_INNER, 0, 0, 0,
      b1, nullptr, nullptr, 0, MTOT, D_INNER, D_MODEL, 1.f, 0);
  gemm_wmma<3, 0><<<gp, 256, 0, stream>>>(hid, D_INNER, 0, 0, 0,
      W2, D_MODEL, 0, 0, 0, y2, D_MODEL, 0, 0, 0,
      b2, ln1f, nullptr, 0, MTOT, D_MODEL, D_INNER, 1.f, 0);

  // ---- 8. LN2 -> d_out
  layernorm_k<<<MTOT, 256, 0, stream>>>(y2, (float*)d_out, nullptr, ln_g, ln_b);
}
